// PeepholeCell_35699768164400
// MI455X (gfx1250) — matrix operand order repaired and checked by reading, not compiled, not run on hardware
//
#include <hip/hip_runtime.h>
#include <hip/hip_bf16.h>
#include <math.h>

// ---------------- problem constants ----------------
constexpr int T = 512, B = 32, I = 512, H = 1024;
constexpr int TB = T * B;                 // 16384 rows of X
constexpr int NBLK = H / 16;              // 64 persistent blocks, 16 output cols each
constexpr int RTHREADS = 256;             // 8 waves per recurrent block

typedef __bf16 bf16_t;
typedef __attribute__((ext_vector_type(16))) __bf16 v16bf;
typedef __attribute__((ext_vector_type(8)))  __bf16 v8bf;
typedef __attribute__((ext_vector_type(8)))  float  v8f;

// gfx1250 async global->LDS path (guarded: falls back to plain copy if absent)
#if defined(__gfx1250__) && __has_builtin(__builtin_amdgcn_global_load_async_to_lds_b128)
#define HAVE_ASYNC_LDS 1
typedef int v4i_vs __attribute__((vector_size(16)));
typedef __attribute__((address_space(1))) void    gas_void;
typedef __attribute__((address_space(3))) void    las_void;
typedef __attribute__((address_space(1))) v4i_vs  gas_v4i;
typedef __attribute__((address_space(3))) v4i_vs  las_v4i;
#else
#define HAVE_ASYNC_LDS 0
#endif

// ---------------- workspace layout (bytes) ----------------
constexpr size_t OFF_BAR  = 0;                          // 2048 u32
constexpr size_t OFF_BIAS = 8192;                       // 4*H f32
constexpr size_t OFF_H    = OFF_BIAS + 4 * H * 4;       // B*H bf16
constexpr size_t OFF_C0   = OFF_H  + (size_t)B * H * 2; // B*H bf16
constexpr size_t OFF_C1   = OFF_C0 + (size_t)B * H * 2; // B*H bf16
constexpr size_t OFF_XBF  = ((OFF_C1 + (size_t)B * H * 2) + 255) & ~size_t(255);
constexpr size_t OFF_WI   = OFF_XBF + (size_t)TB * I * 2;          // 4*H*I bf16
constexpr size_t OFF_WH   = OFF_WI  + (size_t)4 * H * I * 2;       // 7*H*H bf16
constexpr size_t OFF_P    = OFF_WH  + (size_t)7 * H * H * 2;       // 4*TB*H f32

// ---------------- WMMA helpers (bf16 16x16x32, f32 acc) ----------------
__device__ inline v8f wmma_bf16(v16bf a, v16bf b, v8f c) {
  return __builtin_amdgcn_wmma_f32_16x16x32_bf16(
      /*neg_a=*/false, a, /*neg_b=*/false, b,
      /*c_mod=*/(short)0, c, /*reuse_a=*/false, /*reuse_b=*/false);
}

// A-matrix tile 16x32 (MxK), contiguous-K order used here (load_B matches):
//   lanes 0-15: row = lane,   K = 0..15
//   lanes16-31: row = lane-16,K = 16..31
__device__ inline v16bf load_A(const bf16_t* base /*row0,k0*/, int ld, int k0, int lane) {
  int row  = lane & 15;
  int half = lane >> 4;                 // 0 or 1
  const bf16_t* p = base + (size_t)row * ld + k0 + half * 16;
  union { v16bf v; struct { v8bf lo, hi; } s; } u;
  u.s.lo = *(const v8bf*)(p);           // K = half*16 .. +7
  u.s.hi = *(const v8bf*)(p + 8);       // K = half*16+8 .. +15
  return u.v;
}

// B-matrix tile 32x16 (KxN) from W stored row-major [16 cols][ld]:
//   B[k][n] = W[n][k]; lane<16: col=lane, K=0..15; lane>=16: col=lane-16, K=16..31
//   => 16 contiguous K values of row `col` of W: one 32-byte contiguous load.
__device__ inline v16bf load_B(const bf16_t* w, int ld, int k0, int lane) {
  int col  = lane & 15;
  int kofs = (lane >> 4) * 16;
  const bf16_t* p = w + (size_t)col * ld + k0 + kofs;
  return *(const v16bf*)p;
}

__device__ inline float sigm(float x) { return 1.0f / (1.0f + __expf(-x)); }

// ---------------- grid-wide barrier (one fresh slot per event) ----------------
__device__ inline void grid_barrier(unsigned* bar, int slot, unsigned nblk) {
  __syncthreads();
  if (threadIdx.x == 0) {
    __threadfence();
    atomicAdd(&bar[slot], 1u);
    volatile unsigned* p = bar + slot;
    while (*p < nblk) { __builtin_amdgcn_s_sleep(1); }
    __threadfence();
  }
  __syncthreads();
}

// ---------------- prep: bf16 conversion, bias fold, state init ----------------
struct PrepArgs {
  const float* X; const float* h0; const float* c0;
  const float* wi[4];           // w_ii, w_if, w_ic, w_io
  const float* wh[7];           // w_hi, w_ci, w_hf, w_cf, w_hc, w_ho, w_cyo
  const float* bias[11];        // b_ii..b_cyo in dict order
  bf16_t* Xbf; bf16_t* Wi; bf16_t* Wh;
  bf16_t* hbuf; bf16_t* cbuf0;
  float* biasc; unsigned* bar;
};

__global__ void prep_kernel(PrepArgs p) {
  const size_t i      = (size_t)blockIdx.x * blockDim.x + threadIdx.x;
  const size_t stride = (size_t)gridDim.x * blockDim.x;
  constexpr size_t NX = (size_t)TB * I;
  constexpr size_t HI = (size_t)H * I;
  constexpr size_t HH = (size_t)H * H;

  for (size_t j = i; j < NX; j += stride) p.Xbf[j] = (bf16_t)p.X[j];
  for (size_t j = i; j < 4 * HI; j += stride) {
    size_t g = j / HI; p.Wi[j] = (bf16_t)p.wi[g][j - g * HI];
  }
  for (size_t j = i; j < 7 * HH; j += stride) {
    size_t m = j / HH; p.Wh[j] = (bf16_t)p.wh[m][j - m * HH];
  }
  for (size_t j = i; j < (size_t)B * H; j += stride) {
    p.hbuf[j]  = (bf16_t)p.h0[j];
    p.cbuf0[j] = (bf16_t)p.c0[j];
  }
  for (size_t j = i; j < (size_t)H; j += stride) {
    p.biasc[0 * H + j] = p.bias[0][j] + p.bias[1][j] + p.bias[2][j];   // i
    p.biasc[1 * H + j] = p.bias[3][j] + p.bias[4][j] + p.bias[5][j];   // f
    p.biasc[2 * H + j] = p.bias[6][j] + p.bias[7][j];                  // g
    p.biasc[3 * H + j] = p.bias[8][j] + p.bias[9][j] + p.bias[10][j];  // o
  }
  for (size_t j = i; j < 2048; j += stride) p.bar[j] = 0;
}

// ---------------- input projections: P[g,t,b,:] = X @ W_g^T + bias_g ----------------
// one wave per 16x16 D tile; K = I = 512 -> 16 WMMAs per tile
__global__ void xproj_kernel(const bf16_t* __restrict__ Xbf,
                             const bf16_t* __restrict__ Wi,
                             const float*  __restrict__ biasc,
                             float*        __restrict__ P) {
  const int wave = (int)((blockIdx.x * blockDim.x + threadIdx.x) >> 5);
  const int lane = threadIdx.x & 31;
  const int nt = wave & 63;              // 64 N tiles
  const int mt = (wave >> 6) & 1023;     // 1024 M tiles
  const int g  = wave >> 16;             // 4 gates

  const bf16_t* Abase = Xbf + (size_t)mt * 16 * I;
  const bf16_t* Wbase = Wi + (size_t)g * H * I + (size_t)nt * 16 * I;

  v8f acc = {};
  #pragma unroll 4
  for (int k0 = 0; k0 < I; k0 += 32) {
    v16bf a = load_A(Abase, I, k0, lane);
    v16bf b = load_B(Wbase, I, k0, lane);
    acc = wmma_bf16(a, b, acc);
  }

  const int n   = lane & 15;
  const int m0  = (lane >> 4) * 8;       // D layout: VGPR r -> M = m0 + r
  const float bv = biasc[g * H + nt * 16 + n];
  float* outp = P + (size_t)g * TB * H + ((size_t)mt * 16 + m0) * H + nt * 16 + n;
  #pragma unroll
  for (int r = 0; r < 8; ++r) outp[(size_t)r * H] = acc[r] + bv;
}

// ---------------- persistent recurrent kernel ----------------
// 64 blocks; block `blk` owns output columns [16*blk, 16*blk+16).
// LDS: weight slice [7][16][1024] bf16 (224KB) + f32 accum/state (~12KB).
__global__ void recurrent_kernel(const float* __restrict__ c0,
                                 const bf16_t* __restrict__ Wh,
                                 const float* __restrict__ P,
                                 bf16_t* __restrict__ hbuf,
                                 bf16_t* __restrict__ cbuf0,
                                 bf16_t* __restrict__ cbuf1,
                                 unsigned* __restrict__ bar,
                                 float* __restrict__ out) {
  extern __shared__ char smem_raw[];
  bf16_t* sW   = (bf16_t*)smem_raw;                           // 7*16*1024 bf16
  float*  sAcc = (float*)(smem_raw + (size_t)7 * 16 * H * 2); // 4*512 f32
  float*  sC   = sAcc + 4 * 512;                              // 512 f32 (c_t slice)
  float*  sCy  = sC + 512;                                    // 512 f32

  const int tid  = threadIdx.x;
  const int lane = tid & 31;
  const int wave = tid >> 5;
  const int blk  = blockIdx.x;
  const int n0   = blk * 16;

  // --- one-time: this block's weight slice -> LDS (rows n0..n0+15 of 7 matrices)
#if HAVE_ASYNC_LDS
  // async DMA path: 16B per lane per op, tracked by ASYNCcnt (no VGPR data staging)
  for (int idx = tid; idx < 7 * 16 * (H / 8); idx += RTHREADS) {
    int kk  = (idx % (H / 8)) * 8;
    int row = (idx / (H / 8)) % 16;
    int mtx = idx / (16 * (H / 8));
    const bf16_t* src = Wh + ((size_t)mtx * H + n0 + row) * H + kk;
    bf16_t*       dst = sW + ((size_t)mtx * 16 + row) * H + kk;
    gas_v4i* gsrc = (gas_v4i*)(gas_void*)(void*)src;
    las_v4i* ldst = (las_v4i*)(las_void*)(void*)dst;
    __builtin_amdgcn_global_load_async_to_lds_b128(gsrc, ldst, 0, 0);
  }
#if __has_builtin(__builtin_amdgcn_s_wait_asynccnt)
  __builtin_amdgcn_s_wait_asynccnt(0);
#else
  asm volatile("s_wait_asynccnt 0x0" ::: "memory");
#endif
#else
  for (int idx = tid; idx < 7 * 16 * (H / 8); idx += RTHREADS) {
    int kk  = (idx % (H / 8)) * 8;
    int row = (idx / (H / 8)) % 16;
    int mtx = idx / (16 * (H / 8));
    const bf16_t* src = Wh + ((size_t)mtx * H + n0 + row) * H + kk;
    *(v8bf*)(sW + ((size_t)mtx * 16 + row) * H + kk) = *(const v8bf*)src;
  }
#endif
  // --- init c_t slice (f32) from c0
  for (int idx = tid; idx < 512; idx += RTHREADS) {
    int b = idx >> 4, n = idx & 15;
    sC[idx] = c0[b * H + n0 + n];
  }
  __syncthreads();

  const int mrow  = (wave & 1) * 16;     // this wave's batch-row tile (0 or 16)
  const int kbase = (wave >> 1) * 256;   // this wave's K slice of 1024

  for (int t = 0; t < T; ++t) {
    const bf16_t* ccur  = (t & 1) ? cbuf1 : cbuf0;   // c_{t-1}
    bf16_t*       cnext = (t & 1) ? cbuf0 : cbuf1;   // c_t

    // init LDS accumulators with the precomputed projections (bias folded in)
    for (int idx = tid; idx < 512; idx += RTHREADS) {
      int b = idx >> 4, n = idx & 15;
      size_t base = ((size_t)t * B + b) * H + n0 + n;
      sAcc[0 * 512 + idx] = P[0 * (size_t)TB * H + base];
      sAcc[1 * 512 + idx] = P[1 * (size_t)TB * H + base];
      sAcc[2 * 512 + idx] = P[2 * (size_t)TB * H + base];
      sAcc[3 * 512 + idx] = P[3 * (size_t)TB * H + base];
    }
    // prefetch next step's P rows into cache while this step computes
    if (t + 1 < T) {
      for (int idx = tid; idx < B; idx += RTHREADS) {
        size_t base = ((size_t)(t + 1) * B + idx) * H + n0;
        __builtin_prefetch(P + 0 * (size_t)TB * H + base, 0, 1);
        __builtin_prefetch(P + 1 * (size_t)TB * H + base, 0, 1);
        __builtin_prefetch(P + 2 * (size_t)TB * H + base, 0, 1);
        __builtin_prefetch(P + 3 * (size_t)TB * H + base, 0, 1);
      }
    }
    grid_barrier(bar, 2 * t + 0, NBLK);  // h_{t-1}, c_{t-1} bf16 ready everywhere

    // --- phase A: i/f/g matmuls + h-part of o; weights from LDS, h/c from global bf16
    const bf16_t* Ah = hbuf + (size_t)mrow * H;
    const bf16_t* Ac = ccur + (size_t)mrow * H;
    v8f accI = {}, accF = {}, accG = {}, accO = {};
    for (int kk = 0; kk < 256; kk += 32) {
      const int k0 = kbase + kk;
      v16bf ah = load_A(Ah, H, k0, lane);
      v16bf ac = load_A(Ac, H, k0, lane);
      accI = wmma_bf16(ah, load_B(sW + (size_t)0 * 16 * H, H, k0, lane), accI); // w_hi
      accI = wmma_bf16(ac, load_B(sW + (size_t)1 * 16 * H, H, k0, lane), accI); // w_ci
      accF = wmma_bf16(ah, load_B(sW + (size_t)2 * 16 * H, H, k0, lane), accF); // w_hf
      accF = wmma_bf16(ac, load_B(sW + (size_t)3 * 16 * H, H, k0, lane), accF); // w_cf
      accG = wmma_bf16(ah, load_B(sW + (size_t)4 * 16 * H, H, k0, lane), accG); // w_hc
      accO = wmma_bf16(ah, load_B(sW + (size_t)5 * 16 * H, H, k0, lane), accO); // w_ho
    }
    // fold register partials into LDS accumulators (K-split reduction)
    {
      const int n  = lane & 15;
      const int m0 = mrow + (lane >> 4) * 8;
      #pragma unroll
      for (int r = 0; r < 8; ++r) {
        int e = (m0 + r) * 16 + n;
        atomicAdd(&sAcc[0 * 512 + e], accI[r]);
        atomicAdd(&sAcc[1 * 512 + e], accF[r]);
        atomicAdd(&sAcc[2 * 512 + e], accG[r]);
      }
    }
    __syncthreads();

    // --- elementwise: gates i,f,g; new cell state c_t; publish bf16 c_t
    for (int idx = tid; idx < 512; idx += RTHREADS) {
      float iv = sigm(sAcc[0 * 512 + idx]);
      float fv = sigm(sAcc[1 * 512 + idx]);
      float gv = tanhf(sAcc[2 * 512 + idx]);
      float cy = fv * sC[idx] + iv * gv;
      sCy[idx] = cy;
      sC[idx]  = cy;
      int b = idx >> 4, n = idx & 15;
      cnext[b * H + n0 + n] = (bf16_t)cy;
    }
    grid_barrier(bar, 2 * t + 1, NBLK);  // c_t bf16 ready everywhere

    // --- phase B: o-gate peephole on c_t: accO += c_t @ w_cyo^T
    const bf16_t* Acy = cnext + (size_t)mrow * H;
    for (int kk = 0; kk < 256; kk += 32) {
      const int k0 = kbase + kk;
      v16bf acy = load_A(Acy, H, k0, lane);
      accO = wmma_bf16(acy, load_B(sW + (size_t)6 * 16 * H, H, k0, lane), accO); // w_cyo
    }
    {
      const int n  = lane & 15;
      const int m0 = mrow + (lane >> 4) * 8;
      #pragma unroll
      for (int r = 0; r < 8; ++r)
        atomicAdd(&sAcc[3 * 512 + (m0 + r) * 16 + n], accO[r]);
    }
    __syncthreads();

    // --- elementwise: o, h_t; write output f32 and bf16 h for next step
    for (int idx = tid; idx < 512; idx += RTHREADS) {
      float ov = sigm(sAcc[3 * 512 + idx]);
      float hy = ov * tanhf(sCy[idx]);
      int b = idx >> 4, n = idx & 15;
      out[((size_t)t * B + b) * H + n0 + n] = hy;
      hbuf[b * H + n0 + n] = (bf16_t)hy;
    }
    __syncthreads();   // protect sAcc/sCy before next iteration's re-init
  }
}

// ---------------- host entry ----------------
extern "C" void kernel_launch(void* const* d_in, const int* in_sizes, int n_in,
                              void* d_out, int out_size, void* d_ws, size_t ws_size,
                              hipStream_t stream) {
  (void)in_sizes; (void)n_in; (void)out_size; (void)ws_size;
  char* ws = (char*)d_ws;

  unsigned* bar   = (unsigned*)(ws + OFF_BAR);
  float*    biasc = (float*)(ws + OFF_BIAS);
  bf16_t*   hbuf  = (bf16_t*)(ws + OFF_H);
  bf16_t*   cbuf0 = (bf16_t*)(ws + OFF_C0);
  bf16_t*   cbuf1 = (bf16_t*)(ws + OFF_C1);
  bf16_t*   Xbf   = (bf16_t*)(ws + OFF_XBF);
  bf16_t*   Wi    = (bf16_t*)(ws + OFF_WI);
  bf16_t*   Wh    = (bf16_t*)(ws + OFF_WH);
  float*    P     = (float*)(ws + OFF_P);

  PrepArgs p;
  p.X  = (const float*)d_in[0];
  p.h0 = (const float*)d_in[1];
  p.c0 = (const float*)d_in[2];
  p.wi[0] = (const float*)d_in[3];   // w_ii
  p.wi[1] = (const float*)d_in[6];   // w_if
  p.wi[2] = (const float*)d_in[9];   // w_ic
  p.wi[3] = (const float*)d_in[11];  // w_io
  p.wh[0] = (const float*)d_in[4];   // w_hi
  p.wh[1] = (const float*)d_in[5];   // w_ci
  p.wh[2] = (const float*)d_in[7];   // w_hf
  p.wh[3] = (const float*)d_in[8];   // w_cf
  p.wh[4] = (const float*)d_in[10];  // w_hc
  p.wh[5] = (const float*)d_in[12];  // w_ho
  p.wh[6] = (const float*)d_in[13];  // w_cyo
  for (int j = 0; j < 11; ++j) p.bias[j] = (const float*)d_in[14 + j];
  p.Xbf = Xbf; p.Wi = Wi; p.Wh = Wh;
  p.hbuf = hbuf; p.cbuf0 = cbuf0;
  p.biasc = biasc; p.bar = bar;

  prep_kernel<<<2048, 256, 0, stream>>>(p);

  // 4 gates * 1024 M-tiles * 64 N-tiles = 262144 waves; 8 waves/block
  xproj_kernel<<<32768, 256, 0, stream>>>(Xbf, Wi, biasc, P);

  constexpr int SMEM_BYTES = 7 * 16 * H * 2 + (4 * 512 + 512 + 512) * 4; // 241664
  (void)hipFuncSetAttribute((const void*)recurrent_kernel,
                            hipFuncAttributeMaxDynamicSharedMemorySize, SMEM_BYTES);
  recurrent_kernel<<<NBLK, RTHREADS, SMEM_BYTES, stream>>>(
      (const float*)d_in[2], Wh, P, hbuf, cbuf0, cbuf1, bar, (float*)d_out);
}
